// MultiHeadLatentAttention_70171175682544
// MI455X (gfx1250) — matrix operand order repaired and checked by reading, not compiled, not run on hardware
//
#include <hip/hip_runtime.h>
#include <math.h>

// ---------------------------------------------------------------------------
// MLA forward for MI455X (gfx1250, wave32, WMMA).
// All dense math through v_wmma_f32_16x16x32_bf16 (f32 accumulate).
// GEMM: double-buffered LDS + register-staged global loads, packed (v2bf)
// LDS staging for the transposed B tile.
// ---------------------------------------------------------------------------

typedef __bf16 bf16;
typedef __attribute__((ext_vector_type(16))) __bf16 v16bf;
typedef __attribute__((ext_vector_type(8)))  __bf16 v8bf;
typedef __attribute__((ext_vector_type(2)))  __bf16 v2bf;
typedef __attribute__((ext_vector_type(8)))  float  v8f;

#define B_    2
#define S_    2048
#define HID_  2048
#define H_    16
#define D_    128
#define RANK_ 512
#define NEGF  (-1e30f)

// packed pair stores (lower exactly to v_cvt_pk_bf16_f32 + ds_store_b32)
__device__ __forceinline__ void store_pk_f(bf16* dst, float a, float b) {
  v2bf v; v[0] = (bf16)a; v[1] = (bf16)b;
  *(v2bf*)dst = v;
}
__device__ __forceinline__ void store_pk_h(bf16* dst, bf16 a, bf16 b) {
  v2bf v; v[0] = a; v[1] = b;
  *(v2bf*)dst = v;
}

// ---------------- WMMA fragment helpers (wave32 layouts, ISA 7.12.2) -------
// A (16x32 bf16): lane<16 holds row M=lane, elems 0..7 = K0..7, 8..15 = K16..23
//                 lane>=16 holds row M=lane-16, K offsets +8 / +24.
__device__ __forceinline__ v16bf load_a_frag(const bf16* base, int ld) {
  const int lane = threadIdx.x & 31;
  const int hl = lane >> 4;
  const int m  = lane & 15;
  const bf16* p = base + m * ld + hl * 8;
  v8bf lo = *(const v8bf*)(p);
  v8bf hi = *(const v8bf*)(p + 16);
  v16bf r;
#pragma unroll
  for (int i = 0; i < 8; ++i) { r[i] = lo[i]; r[i + 8] = hi[i]; }
  return r;
}

// B (32x16 bf16) staged column-contiguous ([n][k], stride ld):
// lane<16 holds col N=lane, elems 0..7 = K0..7, 8..15 = K16..23; lane>=16 K offsets +8 / +24.
__device__ __forceinline__ v16bf load_b_frag(const bf16* base, int ld) {
  const int lane = threadIdx.x & 31;
  const int hl = lane >> 4;
  const int n  = lane & 15;
  const bf16* p = base + n * ld + hl * 8;
  v8bf lo = *(const v8bf*)(p);
  v8bf hi = *(const v8bf*)(p + 16);
  v16bf r;
#pragma unroll
  for (int i = 0; i < 8; ++i) { r[i] = lo[i]; r[i + 8] = hi[i]; }
  return r;
}

__device__ __forceinline__ v8f wmma_bf16(v16bf a, v16bf b, v8f c) {
  return __builtin_amdgcn_wmma_f32_16x16x32_bf16(false, a, false, b,
                                                 (short)0, c, false, false);
}

// ---------------- Generic GEMM: C[MxN] = A[MxK] * W[KxN] (fp32 io) ---------
// Block tile 128x128, K-step 32, 256 threads = 8 waves (4 Mgroups x 2 Ngroups),
// each wave owns 2x4 = 8 accumulator tiles of 16x16.
// Double-buffered LDS; next K-tile is register-staged while WMMAs run.
__global__ __launch_bounds__(256) void gemm_bf16_wmma(
    const float* __restrict__ A, const float* __restrict__ W,
    float* __restrict__ C, int K, int lda, int ldb, int ldc) {
  __shared__ __align__(16) bf16 sA[2][128 * 32];   // [m][k]
  __shared__ __align__(16) bf16 sB[2][128 * 32];   // [n][k] (transposed)

  const int tid  = threadIdx.x;
  const int lane = tid & 31;
  const int wid  = tid >> 5;
  const int wm   = wid & 3;         // 0..3 -> 32 rows each
  const int wn   = wid >> 2;        // 0..1 -> 64 cols each
  const int hl   = lane >> 4;
  const int nl   = lane & 15;
  const int m0   = blockIdx.y * 128;
  const int n0   = blockIdx.x * 128;

  // staging positions
  const int ra = tid >> 3;              // A: 0..31 row group (8 lanes per row)
  const int ca = (tid & 7) * 4;         // A: 0,4,..,28 col
  const int kB = (tid >> 4) * 2;        // B: even k row 0,2,..,30
  const int nB = (tid & 15) * 4;        // B: 0,4,..,60 (pass adds +64)

  v8f acc[2][4];
#pragma unroll
  for (int i = 0; i < 2; ++i)
#pragma unroll
    for (int j = 0; j < 4; ++j)
#pragma unroll
      for (int e = 0; e < 8; ++e) acc[i][j][e] = 0.0f;

  float4 aReg[4], b0Reg[2], b1Reg[2];
  auto load_tiles = [&](int k0) {
#pragma unroll
    for (int p = 0; p < 4; ++p) {
      const float* pa = A + (size_t)(m0 + ra + p * 32) * lda + k0 + ca;
      aReg[p] = *(const float4*)pa;
      __builtin_prefetch(pa + 64, 0, 1);  // speculative, next-next K tile
    }
#pragma unroll
    for (int p = 0; p < 2; ++p) {
      const float* pb0 = W + (size_t)(k0 + kB) * ldb + n0 + nB + p * 64;
      const float* pb1 = pb0 + ldb;
      b0Reg[p] = *(const float4*)pb0;
      b1Reg[p] = *(const float4*)pb1;
      __builtin_prefetch(pb0 + (size_t)32 * ldb, 0, 1);
    }
  };
  auto store_tiles = [&](int buf) {
#pragma unroll
    for (int p = 0; p < 4; ++p) {
      bf16* d = &sA[buf][(ra + p * 32) * 32 + ca];
      d[0] = (bf16)aReg[p].x; d[1] = (bf16)aReg[p].y;
      d[2] = (bf16)aReg[p].z; d[3] = (bf16)aReg[p].w;
    }
    // transposed B: (n, k) and (n, k+1) are contiguous -> packed pair stores
#pragma unroll
    for (int p = 0; p < 2; ++p) {
      const int nb = nB + p * 64;
      store_pk_f(&sB[buf][(nb + 0) * 32 + kB], b0Reg[p].x, b1Reg[p].x);
      store_pk_f(&sB[buf][(nb + 1) * 32 + kB], b0Reg[p].y, b1Reg[p].y);
      store_pk_f(&sB[buf][(nb + 2) * 32 + kB], b0Reg[p].z, b1Reg[p].z);
      store_pk_f(&sB[buf][(nb + 3) * 32 + kB], b0Reg[p].w, b1Reg[p].w);
    }
  };

  load_tiles(0);
  int buf = 0;
  for (int k0 = 0; k0 < K; k0 += 32) {
    store_tiles(buf);
    __syncthreads();
    if (k0 + 32 < K) load_tiles(k0 + 32);   // overlap with WMMAs below

    v16bf af[2], bfr[4];
#pragma unroll
    for (int i = 0; i < 2; ++i)
      af[i] = load_a_frag(&sA[buf][(wm * 32 + i * 16) * 32], 32);
#pragma unroll
    for (int j = 0; j < 4; ++j)
      bfr[j] = load_b_frag(&sB[buf][(wn * 64 + j * 16) * 32], 32);
#pragma unroll
    for (int i = 0; i < 2; ++i)
#pragma unroll
      for (int j = 0; j < 4; ++j)
        acc[i][j] = wmma_bf16(af[i], bfr[j], acc[i][j]);
    buf ^= 1;
  }

  // write back (C/D layout: elem r -> row r + 8*hl, col = nl)
#pragma unroll
  for (int i = 0; i < 2; ++i) {
    const int rowb = m0 + wm * 32 + i * 16 + hl * 8;
#pragma unroll
    for (int j = 0; j < 4; ++j) {
      const int col = n0 + wn * 64 + j * 16 + nl;
#pragma unroll
      for (int r = 0; r < 8; ++r)
        C[(size_t)(rowb + r) * ldc + col] = acc[i][j][r];
    }
  }
}

// ---------------- RMSNorm over first 512 cols of kv_a (eps 1e-6) -----------
__global__ __launch_bounds__(256) void rmsnorm_kernel(
    const float* __restrict__ x, const float* __restrict__ w,
    float* __restrict__ y) {
  const int row = blockIdx.x;                 // 0..B*S-1
  const float* xr = x + (size_t)row * (RANK_ + D_);
  __shared__ float red[8];
  float ss = 0.0f;
  for (int i = threadIdx.x; i < RANK_; i += 256) {
    const float v = xr[i];
    ss += v * v;
  }
#pragma unroll
  for (int off = 16; off; off >>= 1) ss += __shfl_xor(ss, off, 32);
  if ((threadIdx.x & 31) == 0) red[threadIdx.x >> 5] = ss;
  __syncthreads();
  if (threadIdx.x == 0) {
    float t = 0.0f;
#pragma unroll
    for (int i = 0; i < 8; ++i) t += red[i];
    red[0] = t;
  }
  __syncthreads();
  const float rn = rsqrtf(red[0] / (float)RANK_ + 1e-6f);
  for (int i = threadIdx.x; i < RANK_; i += 256)
    y[(size_t)row * RANK_ + i] = w[i] * xr[i] * rn;
}

// ---------------- YaRN RoPE + pack q/k/v into bf16 (b,h,s,d) ---------------
__global__ __launch_bounds__(128) void rope_pack_kernel(
    const float* __restrict__ q_flat, const float* __restrict__ kva,
    const float* __restrict__ kv, const int* __restrict__ pos_ids,
    bf16* __restrict__ qb, bf16* __restrict__ kb, bf16* __restrict__ vb) {
  const int s = blockIdx.x, b = blockIdx.y;
  const int d = threadIdx.x;                  // 0..127
  const int row = b * S_ + s;
  const float pos = (float)pos_ids[row];

  // YaRN inv_freq
  const int i = d & 63;
  const float PIf = 3.14159265358979323846f;
  const float pf = powf(500000.0f, (float)i / 64.0f);
  const float extrap = 1.0f / pf;
  const float interp = 1.0f / (16.0f * pf);
  const float lnb2 = 2.0f * logf(500000.0f);
  float low  = floorf(128.0f * logf(8192.0f / (32.0f * 2.0f * PIf)) / lnb2);
  float high = ceilf(128.0f * logf(8192.0f / (2.0f * PIf)) / lnb2);
  low = fmaxf(low, 0.0f);
  high = fminf(high, 127.0f);
  float sm = ((float)i - low) / fmaxf(high - low, 1.0f);
  sm = fminf(fmaxf(sm, 0.0f), 1.0f);
  const float invf = (1.0f - sm) * interp + sm * extrap;
  const float mscale = 0.1f * logf(16.0f) + 1.0f;
  const float ang = pos * invf;
  const float c  = cosf(ang) * mscale;
  const float sn = sinf(ang) * mscale;

  // rotate_half index
  const int dro = (d < 64) ? d + 64 : d - 64;
  const float sgn = (d < 64) ? -1.0f : 1.0f;

  // k_rope (single head, broadcast)
  const float kr  = kva[(size_t)row * (RANK_ + D_) + RANK_ + d];
  const float kro = kva[(size_t)row * (RANK_ + D_) + RANK_ + dro];
  const float krot = kr * c + sgn * kro * sn;

  const float qscale = 0.08838834764831845f;  // 1/sqrt(128)

#pragma unroll
  for (int h = 0; h < H_; ++h) {
    const float qv = q_flat[(size_t)row * HID_ + h * D_ + d];
    const float qo = q_flat[(size_t)row * HID_ + h * D_ + dro];
    const float qr = qv * c + sgn * qo * sn;
    const size_t o = ((size_t)(b * H_ + h) * S_ + s) * D_ + d;
    qb[o] = (bf16)(qr * qscale);
    const float kn = kv[(size_t)row * (2 * H_ * D_) + h * 256 + d];
    kb[o] = (bf16)(kn + krot);
    vb[o] = (bf16)kv[(size_t)row * (2 * H_ * D_) + h * 256 + 128 + d];
  }
}

// ---------------- Flash attention (causal), WMMA for QK^T and PV -----------
// Block = (128 q rows, one (b,h)). 8 waves, 16 q rows each. KV tiles of 64.
__global__ __launch_bounds__(256) void mla_attn_kernel(
    const bf16* __restrict__ qb, const bf16* __restrict__ kb,
    const bf16* __restrict__ vb, float* __restrict__ attn) {
  __shared__ __align__(16) bf16 sK[64 * 128];      // [kv][d]
  __shared__ __align__(16) bf16 sVt[128 * 64];     // [d][kv]
  __shared__ __align__(16) bf16 sP[8][16 * 64];    // per-wave [m][kv]

  const int qt = blockIdx.x, h = blockIdx.y, b = blockIdx.z;
  const int q0 = qt * 128;
  const int tid = threadIdx.x, lane = tid & 31, wid = tid >> 5;
  const int hl = lane >> 4, nl = lane & 15;

  const bf16* qg = qb + ((size_t)(b * H_ + h) * S_ + q0 + wid * 16) * D_;
  const bf16* kg = kb + ((size_t)(b * H_ + h) * S_) * D_;
  const bf16* vg = vb + ((size_t)(b * H_ + h) * S_) * D_;

  // preload Q fragments (scale already folded in)
  v16bf qa[4];
#pragma unroll
  for (int ks = 0; ks < 4; ++ks) qa[ks] = load_a_frag(qg + ks * 32, D_);

  v8f o[8];
#pragma unroll
  for (int dt = 0; dt < 8; ++dt)
#pragma unroll
    for (int e = 0; e < 8; ++e) o[dt][e] = 0.0f;
  float mrow[8], se[8];
#pragma unroll
  for (int r = 0; r < 8; ++r) { mrow[r] = NEGF; se[r] = 0.0f; }

  const int tmax = (q0 + 127) >> 6;
  for (int t = 0; t <= tmax; ++t) {
    const int kv0 = t * 64;
    // stage K tile (straight 16B copies)
    {
      const uint4* src = (const uint4*)(kg + (size_t)kv0 * D_);
      uint4* dst = (uint4*)sK;
#pragma unroll
      for (int i = 0; i < 4; ++i) dst[tid + i * 256] = src[tid + i * 256];
    }
    // stage V tile transposed: vector global loads of kv-pairs, packed b32
    // scatter into [d][kv] (kv, kv+1 contiguous).
    {
      const bf16* vsrc = vg + (size_t)kv0 * D_;
      const int d8 = (tid & 15) * 8;
#pragma unroll
      for (int p = 0; p < 2; ++p) {
        const int kv2 = ((tid >> 4) + p * 16) * 2;   // 0,2,..,62
        const v8bf va = *(const v8bf*)(vsrc + (size_t)kv2 * D_ + d8);
        const v8bf vbv = *(const v8bf*)(vsrc + (size_t)(kv2 + 1) * D_ + d8);
#pragma unroll
        for (int j = 0; j < 8; ++j)
          store_pk_h(&sVt[(d8 + j) * 64 + kv2], va[j], vbv[j]);
      }
    }
    __syncthreads();

    // S = Q * K^T  (16 x 64 per wave)
    v8f sfr[4];
#pragma unroll
    for (int n = 0; n < 4; ++n) {
#pragma unroll
      for (int e = 0; e < 8; ++e) sfr[n][e] = 0.0f;
#pragma unroll
      for (int ks = 0; ks < 4; ++ks) {
        const v16bf bf = load_b_frag(sK + (n * 16) * 128 + ks * 32, 128);
        sfr[n] = wmma_bf16(qa[ks], bf, sfr[n]);
      }
    }

    // causal mask + online softmax (half-wave shuffle reductions)
#pragma unroll
    for (int r = 0; r < 8; ++r) {
      const int qrow = q0 + wid * 16 + r + 8 * hl;
      float mx = NEGF;
#pragma unroll
      for (int n = 0; n < 4; ++n) {
        const int kvp = kv0 + n * 16 + nl;
        float v = sfr[n][r];
        if (kvp > qrow) v = NEGF;
        sfr[n][r] = v;
        mx = fmaxf(mx, v);
      }
#pragma unroll
      for (int off = 1; off < 16; off <<= 1) mx = fmaxf(mx, __shfl_xor(mx, off, 32));
      const float mnew = fmaxf(mrow[r], mx);
      const float corr = __expf(mrow[r] - mnew);
      float ls = 0.0f;
#pragma unroll
      for (int n = 0; n < 4; ++n) {
        const float p = __expf(sfr[n][r] - mnew);
        sfr[n][r] = p;
        ls += p;
      }
#pragma unroll
      for (int off = 1; off < 16; off <<= 1) ls += __shfl_xor(ls, off, 32);
      se[r] = se[r] * corr + ls;
      mrow[r] = mnew;
#pragma unroll
      for (int dt = 0; dt < 8; ++dt) o[dt][r] *= corr;
      // transpose P through LDS (C layout -> A layout)
#pragma unroll
      for (int n = 0; n < 4; ++n)
        sP[wid][(r + 8 * hl) * 64 + n * 16 + nl] = (bf16)sfr[n][r];
    }
    __syncthreads();

    // O += P * V  (16 x 128 per wave)
#pragma unroll
    for (int ks = 0; ks < 2; ++ks) {
      const v16bf pa = load_a_frag(&sP[wid][0] + ks * 32, 64);
#pragma unroll
      for (int dt = 0; dt < 8; ++dt) {
        const v16bf bv = load_b_frag(sVt + (dt * 16) * 64 + ks * 32, 64);
        o[dt] = wmma_bf16(pa, bv, o[dt]);
      }
    }
    __syncthreads();
  }

  // epilogue: out = o / (se + EPS), layout (b, s, h*D)
#pragma unroll
  for (int r = 0; r < 8; ++r) {
    const float inv = 1.0f / (se[r] + 1e-5f);
    const int srow = q0 + wid * 16 + r + 8 * hl;
    float* orow = attn + ((size_t)(b * S_ + srow)) * (H_ * D_) + h * D_;
#pragma unroll
    for (int dt = 0; dt < 8; ++dt)
      orow[dt * 16 + nl] = o[dt][r] * inv;
  }
}

// ---------------------------------------------------------------------------
extern "C" void kernel_launch(void* const* d_in, const int* in_sizes, int n_in,
                              void* d_out, int out_size, void* d_ws, size_t ws_size,
                              hipStream_t stream) {
  (void)in_sizes; (void)n_in; (void)out_size; (void)ws_size;
  const float* hidden  = (const float*)d_in[0];
  const int*   pos_ids = (const int*)d_in[1];
  const float* w_q     = (const float*)d_in[2];
  const float* w_kv_a  = (const float*)d_in[3];
  const float* w_kv_b  = (const float*)d_in[4];
  const float* w_o     = (const float*)d_in[5];
  const float* kv_ln   = (const float*)d_in[6];
  float* out = (float*)d_out;

  const size_t ROWS = (size_t)B_ * S_;                 // 4096
  char* ws = (char*)d_ws;
  size_t off = 0;
  auto take = [&](size_t bytes) -> void* {
    void* p = ws + off;
    off += (bytes + 255) & ~(size_t)255;
    return p;
  };
  float* q_flat = (float*)take(ROWS * HID_ * sizeof(float));           // (b,s,h*d)
  float* kva    = (float*)take(ROWS * (RANK_ + D_) * sizeof(float));
  float* ckv    = (float*)take(ROWS * RANK_ * sizeof(float));
  float* kvb    = (float*)take(ROWS * (2 * H_ * D_) * sizeof(float));  // (b,s,h,2d)
  bf16*  qbuf   = (bf16*)take((size_t)B_ * H_ * S_ * D_ * sizeof(bf16));
  bf16*  kbuf   = (bf16*)take((size_t)B_ * H_ * S_ * D_ * sizeof(bf16));
  bf16*  vbuf   = (bf16*)take((size_t)B_ * H_ * S_ * D_ * sizeof(bf16));
  float* attn   = (float*)take(ROWS * HID_ * sizeof(float));

  const dim3 blk256(256);

  // q = hidden @ w_q              (4096 x 2048 x 2048)
  gemm_bf16_wmma<<<dim3(HID_ / 128, ROWS / 128), blk256, 0, stream>>>(
      hidden, w_q, q_flat, HID_, HID_, HID_, HID_);
  // kv_a = hidden @ w_kv_a        (4096 x 640 x 2048)
  gemm_bf16_wmma<<<dim3((RANK_ + D_) / 128, ROWS / 128), blk256, 0, stream>>>(
      hidden, w_kv_a, kva, HID_, HID_, RANK_ + D_, RANK_ + D_);
  // c_kv = RMSNorm(kv_a[:, :512])
  rmsnorm_kernel<<<dim3((unsigned)ROWS), blk256, 0, stream>>>(kva, kv_ln, ckv);
  // kv = c_kv @ w_kv_b            (4096 x 4096 x 512)
  gemm_bf16_wmma<<<dim3((2 * H_ * D_) / 128, ROWS / 128), blk256, 0, stream>>>(
      ckv, w_kv_b, kvb, RANK_, RANK_, 2 * H_ * D_, 2 * H_ * D_);
  // RoPE + pack to bf16 (b,h,s,d); q pre-scaled by 1/sqrt(D)
  rope_pack_kernel<<<dim3(S_, B_), dim3(128), 0, stream>>>(
      q_flat, kva, kvb, pos_ids, qbuf, kbuf, vbuf);
  // causal flash attention
  mla_attn_kernel<<<dim3(S_ / 128, H_, B_), blk256, 0, stream>>>(
      qbuf, kbuf, vbuf, attn);
  // out = attn @ w_o              (4096 x 2048 x 2048)
  gemm_bf16_wmma<<<dim3(HID_ / 128, ROWS / 128), blk256, 0, stream>>>(
      attn, w_o, out, HID_, HID_, HID_, HID_);
}
